// LieAlgebraRelativeAttention_70196945485879
// MI455X (gfx1250) — matrix operand order repaired and checked by reading, not compiled, not run on hardware
//
#include <hip/hip_runtime.h>

typedef __bf16 bf16;
typedef __attribute__((ext_vector_type(16))) __bf16 v16bf;
typedef __attribute__((ext_vector_type(8)))  __bf16 v8bf;
typedef __attribute__((ext_vector_type(8)))  float  v8f;
typedef __attribute__((ext_vector_type(4)))  unsigned int v4u;
typedef __attribute__((ext_vector_type(4)))  int v4i;

#define B_  2
#define H_  4
#define T_  2048
#define D_  63
#define DP  64      // padded head dim
#define S_  21      // sub-vectors
#define NT  (T_/64) // 32 key tiles

#if __has_builtin(__builtin_amdgcn_global_load_async_to_lds_b128) && \
    __has_builtin(__builtin_amdgcn_s_wait_asynccnt)
#define USE_ASYNC 1
#else
#define USE_ASYNC 0
#endif

// builtin expects typed pointers: (addrspace(1) v4i*, addrspace(3) v4i*, Ii, Ii)
#define AS1_V4I(p) ((__attribute__((address_space(1))) v4i*)((void*)(p)))
#define AS3_V4I(p) ((__attribute__((address_space(3))) v4i*)((void*)(p)))

// ---------------------------------------------------------------------------
// Kernel 1: quaternion Lie-algebra rotation of q,k + bf16 conversion (padded).
// One thread per (b,h,t,subvector). 1/sqrt(63) folded into q_rot.
// V is written TRANSPOSED per 64-row tile: vt[bh][t/64][d][t%64], so that each
// (64 x 64) K tile and V^T tile is a contiguous 8KB block -> linear async DMA.
// ---------------------------------------------------------------------------
__global__ void geope_rotate_kernel(const float* __restrict__ q,
                                    const float* __restrict__ k,
                                    const float* __restrict__ v,
                                    const float* __restrict__ coords,
                                    bf16* __restrict__ qrot,
                                    bf16* __restrict__ krot,
                                    bf16* __restrict__ vt)
{
    int i = blockIdx.x * blockDim.x + threadIdx.x;
    const int total = B_ * H_ * T_ * S_;
    if (i >= total) return;

    int s = i % S_;
    int t = (i / S_) % T_;
    int h = (i / (S_ * T_)) % H_;
    int b =  i / (S_ * T_ * H_);

    const float cs = 0.0001f;
    const float* cp = coords + ((size_t)b * T_ + t) * 3;
    float c0 = cp[0] * cs, c1 = cp[1] * cs, c2 = cp[2] * cs;

    // inv_freq = 32^(-s/21) = 2^(-5s/21)
    float invf = exp2f(-5.0f * (float)s / 21.0f);
    float tx = c0 * invf, ty = c1 * invf, tz = c2 * invf;

    float Theta = (1.0f / 3.0f) * sqrtf(tx * tx + ty * ty + tz * tz + 1e-8f);
    float half  = 0.5f * Theta;
    float w     = __cosf(half);
    float sc    = __sinf(half) / (3.0f * Theta + 1e-8f);
    float qx = sc * tx, qy = sc * ty, qz = sc * tz;

    size_t bht     = (size_t)(b * H_ + h) * T_ + t;
    size_t in_off  = bht * D_ + (size_t)s * 3;
    size_t out_off = bht * DP + (size_t)s * 3;

    const float qscale = 0.12598815766974242f; // 1/sqrt(63)

    // rotate query (row-major, padded)
    {
        float vx = q[in_off + 0], vy = q[in_off + 1], vz = q[in_off + 2];
        float t0 = qy * vz - qz * vy;
        float t1 = qz * vx - qx * vz;
        float t2 = qx * vy - qy * vx;
        float px = vx + 2.f * w * t0 + 2.f * (qy * t2 - qz * t1);
        float py = vy + 2.f * w * t1 + 2.f * (qz * t0 - qx * t2);
        float pz = vz + 2.f * w * t2 + 2.f * (qx * t1 - qy * t0);
        qrot[out_off + 0] = (bf16)(px * qscale);
        qrot[out_off + 1] = (bf16)(py * qscale);
        qrot[out_off + 2] = (bf16)(pz * qscale);
    }
    // rotate key (row-major, padded)
    {
        float vx = k[in_off + 0], vy = k[in_off + 1], vz = k[in_off + 2];
        float t0 = qy * vz - qz * vy;
        float t1 = qz * vx - qx * vz;
        float t2 = qx * vy - qy * vx;
        float px = vx + 2.f * w * t0 + 2.f * (qy * t2 - qz * t1);
        float py = vy + 2.f * w * t1 + 2.f * (qz * t0 - qx * t2);
        float pz = vz + 2.f * w * t2 + 2.f * (qx * t1 - qy * t0);
        krot[out_off + 0] = (bf16)px;
        krot[out_off + 1] = (bf16)py;
        krot[out_off + 2] = (bf16)pz;
    }
    // value: convert + per-tile transpose  vt[bh][t/64][d][t%64]
    {
        size_t vbase = (((size_t)(b * H_ + h) * NT + (t >> 6)) * DP) * 64 + (t & 63);
        vt[vbase + (size_t)(s * 3 + 0) * 64] = (bf16)v[in_off + 0];
        vt[vbase + (size_t)(s * 3 + 1) * 64] = (bf16)v[in_off + 1];
        vt[vbase + (size_t)(s * 3 + 2) * 64] = (bf16)v[in_off + 2];
        if (s == 0) vt[vbase + (size_t)63 * 64] = (bf16)0.0f;
    }

    if (s == 0) { // zero the pad column d=63 for q,k
        qrot[bht * DP + 63] = (bf16)0.0f;
        krot[bht * DP + 63] = (bf16)0.0f;
    }
}

// ---------------------------------------------------------------------------
// Stage one contiguous 8KB tile (4096 bf16) global -> LDS.
// Async path: 4 GLOBAL_LOAD_ASYNC_TO_LDS_B128 per thread (each wave-instr
// moves 32 lanes x 16B = 512B, ASYNCcnt-tracked). Fallback: vector copy.
// ---------------------------------------------------------------------------
__device__ __forceinline__ void tile_stage(bf16* lds, const bf16* gsrc, int tid)
{
#pragma unroll
    for (int i = 0; i < 4; ++i) {
        int c = tid + i * 128;          // 512 chunks of 16B
#if USE_ASYNC
        __builtin_amdgcn_global_load_async_to_lds_b128(
            AS1_V4I(gsrc + c * 8), AS3_V4I(lds + c * 8), 0, 0);
#else
        *(v4u*)(lds + c * 8) = *(const v4u*)(gsrc + c * 8);
#endif
    }
}

// ---------------------------------------------------------------------------
// Kernel 2: flash attention on v_wmma_f32_16x16x32_bf16.
// Block = 128 threads (4 waves); wave owns 16 query rows; block covers 64.
// K and V^T tiles double-buffered in LDS, filled by async DMA overlapping
// the previous tile's WMMA + online-softmax compute.
//
// WMMA per-lane layouts (wave32, CDNA5 ISA 7.12.2):
//   A (16x32 bf16): lane L: m=L&15, hk=L>>4; elems 0..7 = K(hk*8+e),
//                   elems 8..15 = K(16+hk*8+e)               -> 2x16B loads
//   B (32x16 bf16): lane L: n=L&15; same K order as A         -> 2x16B loads
//   C/D (16x16 f32): reg r: row = r + 8*(L>>4), col = L&15
// ---------------------------------------------------------------------------
__global__ __launch_bounds__(128) void flash_attn_kernel(
    const bf16* __restrict__ qrot,
    const bf16* __restrict__ krot,
    const bf16* __restrict__ vt,
    float* __restrict__ out)
{
    __shared__ __align__(16) bf16 KtB[2][DP * 64];    // K tile, row-major [j][d], 2x8KB
    __shared__ __align__(16) bf16 VtB[2][DP * 64];    // V tile transposed [d][j], 2x8KB
    __shared__ __align__(16) bf16 Pbuf[4][16 * 64];   // per-wave P staging, 8KB

    const int tid  = threadIdx.x;
    const int lane = tid & 31;
    const int warp = tid >> 5;
    const int bh   = blockIdx.y;
    const int q0   = blockIdx.x * 64 + warp * 16;

    const int m  = lane & 15;
    const int hk = lane >> 4;

    // ---- load Q A-fragments (live in registers across the whole loop) ----
    const bf16* qbase = qrot + ((size_t)bh * T_ + q0 + m) * DP;
    v16bf aq[2];
#pragma unroll
    for (int half = 0; half < 2; ++half) {
        v8bf lo = *(const v8bf*)(qbase + half * 32 + hk * 8);
        v8bf hi = *(const v8bf*)(qbase + half * 32 + 16 + hk * 8);
#pragma unroll
        for (int e = 0; e < 8; ++e) { aq[half][e] = lo[e]; aq[half][8 + e] = hi[e]; }
    }

    v8f   acc[4];
    float mrow[8], lrow[8];
#pragma unroll
    for (int dc = 0; dc < 4; ++dc)
#pragma unroll
        for (int r = 0; r < 8; ++r) acc[dc][r] = 0.0f;
#pragma unroll
    for (int r = 0; r < 8; ++r) { mrow[r] = -INFINITY; lrow[r] = 0.0f; }

    bf16* pb = &Pbuf[warp][0];
    const bf16* kroot = krot + (size_t)bh * T_ * DP;
    const bf16* vroot = vt   + (size_t)bh * T_ * DP;   // tile jt at offset jt*64*DP

    // prologue: stage tile 0 into buffer 0
    tile_stage(KtB[0], kroot, tid);
    tile_stage(VtB[0], vroot, tid);

    for (int jt = 0; jt < NT; ++jt) {
        const int buf = jt & 1;
#if USE_ASYNC
        __builtin_amdgcn_s_wait_asynccnt(0);
#endif
        __syncthreads();   // tile jt visible to all; previous reads of buf^1 done

        // kick off DMA for the next tile into the other buffer
        if (jt + 1 < NT) {
            tile_stage(KtB[buf ^ 1], kroot + (size_t)(jt + 1) * 64 * DP, tid);
            tile_stage(VtB[buf ^ 1], vroot + (size_t)(jt + 1) * 64 * DP, tid);
        }

        const bf16* Kt = KtB[buf];
        const bf16* Vt = VtB[buf];

        // ---- S = Q_rot K_rot^T (scaled): 4 N-chunks x 2 K-halves ----
        v8f s[4];
#pragma unroll
        for (int nc = 0; nc < 4; ++nc) {
            v8f c;
#pragma unroll
            for (int r = 0; r < 8; ++r) c[r] = 0.0f;
#pragma unroll
            for (int half = 0; half < 2; ++half) {
                const bf16* kr = Kt + (nc * 16 + m) * DP + half * 32;
                v8bf klo = *(const v8bf*)(kr + hk * 8);
                v8bf khi = *(const v8bf*)(kr + 16 + hk * 8);
                v16bf bk;
#pragma unroll
                for (int e = 0; e < 8; ++e) { bk[e] = klo[e]; bk[8 + e] = khi[e]; }
                c = __builtin_amdgcn_wmma_f32_16x16x32_bf16(
                        false, aq[half], false, bk, (short)0, c, false, false);
            }
            s[nc] = c;
        }

        // ---- online softmax: row max / rescale / exp / row sum ----
        float alpha[8];
#pragma unroll
        for (int r = 0; r < 8; ++r) {
            float mx = fmaxf(fmaxf(s[0][r], s[1][r]), fmaxf(s[2][r], s[3][r]));
#pragma unroll
            for (int msk = 1; msk < 16; msk <<= 1)
                mx = fmaxf(mx, __shfl_xor(mx, msk, 16));
            float mnew = fmaxf(mrow[r], mx);
            alpha[r]   = __expf(mrow[r] - mnew);
            mrow[r]    = mnew;
        }

#pragma unroll
        for (int r = 0; r < 8; ++r) {
            float rs = 0.0f;
#pragma unroll
            for (int nc = 0; nc < 4; ++nc) {
                float p = __expf(s[nc][r] - mrow[r]);
                rs += p;
                pb[(r + hk * 8) * 64 + nc * 16 + m] = (bf16)p;  // D-layout -> row-major
            }
#pragma unroll
            for (int msk = 1; msk < 16; msk <<= 1)
                rs += __shfl_xor(rs, msk, 16);
            lrow[r] = lrow[r] * alpha[r] + rs;
#pragma unroll
            for (int dc = 0; dc < 4; ++dc) acc[dc][r] *= alpha[r];
        }

        // ---- reload P in A-layout (per-wave LDS, wave-internal ordering) ----
        v16bf ap[2];
#pragma unroll
        for (int half = 0; half < 2; ++half) {
            const bf16* pr = pb + m * 64 + half * 32;
            v8bf lo = *(const v8bf*)(pr + hk * 8);
            v8bf hi = *(const v8bf*)(pr + 16 + hk * 8);
#pragma unroll
            for (int e = 0; e < 8; ++e) { ap[half][e] = lo[e]; ap[half][8 + e] = hi[e]; }
        }

        // ---- O += P V : 4 d-chunks x 2 j-halves ----
#pragma unroll
        for (int dc = 0; dc < 4; ++dc) {
            v8f c = acc[dc];
#pragma unroll
            for (int half = 0; half < 2; ++half) {
                const bf16* vr = Vt + (dc * 16 + m) * 64 + half * 32;
                v8bf vlo = *(const v8bf*)(vr + hk * 8);
                v8bf vhi = *(const v8bf*)(vr + 16 + hk * 8);
                v16bf bv;
#pragma unroll
                for (int e = 0; e < 8; ++e) { bv[e] = vlo[e]; bv[8 + e] = vhi[e]; }
                c = __builtin_amdgcn_wmma_f32_16x16x32_bf16(
                        false, ap[half], false, bv, (short)0, c, false, false);
            }
            acc[dc] = c;
        }
    }

    // ---- epilogue: normalize and store fp32 output (skip pad col 63) ----
    float* obase = out + (size_t)bh * T_ * D_;
#pragma unroll
    for (int dc = 0; dc < 4; ++dc) {
        int col = dc * 16 + m;
#pragma unroll
        for (int r = 0; r < 8; ++r) {
            int row = q0 + r + hk * 8;
            float val = acc[dc][r] / lrow[r];
            if (col < D_) obase[(size_t)row * D_ + col] = val;
        }
    }
}

// ---------------------------------------------------------------------------
extern "C" void kernel_launch(void* const* d_in, const int* in_sizes, int n_in,
                              void* d_out, int out_size, void* d_ws, size_t ws_size,
                              hipStream_t stream)
{
    const float* q      = (const float*)d_in[0];
    const float* k      = (const float*)d_in[1];
    const float* v      = (const float*)d_in[2];
    const float* coords = (const float*)d_in[3];

    const size_t nelem = (size_t)B_ * H_ * T_ * DP;   // padded bf16 tensor size
    bf16* qrot = (bf16*)d_ws;
    bf16* krot = qrot + nelem;
    bf16* vt   = krot + nelem;

    int total = B_ * H_ * T_ * S_;
    geope_rotate_kernel<<<(total + 255) / 256, 256, 0, stream>>>(
        q, k, v, coords, qrot, krot, vt);

    dim3 grid(T_ / 64, B_ * H_);
    flash_attn_kernel<<<grid, 128, 0, stream>>>(qrot, krot, vt, (float*)d_out);
}
